// FrequencyDomainModulatorV2_67181878444188
// MI455X (gfx1250) — matrix operand order repaired and checked by reading, not compiled, not run on hardware
//
#include <hip/hip_runtime.h>
#include <math.h>

// ---------------------------------------------------------------------------
// FrequencyDomainModulatorV2 for MI455X (gfx1250, wave32, WMMA bf16)
//
// Whole pipeline expressed as bf16 WMMA GEMMs with f32 accumulation:
//   DFT rows -> DFT cols -> (Q,K conv + elu + S=QK^T) -> M_b = wo*S_bd*wv ->
//   op = M_b*fc -> iDFT rows -> iDFT cols -> ws conv + sigmoid
// ---------------------------------------------------------------------------

typedef __bf16 bf16_t;
typedef __attribute__((ext_vector_type(16))) __bf16 v16bf;
typedef __attribute__((ext_vector_type(8)))  float  v8f;

__device__ __forceinline__ v8f wmma_bf16(v16bf a, v16bf b, v8f c) {
  return __builtin_amdgcn_wmma_f32_16x16x32_bf16(false, a, false, b, (short)0, c,
                                                 false, false);
}

__device__ __forceinline__ bf16_t f2bf(float f) {
  union { float f; unsigned u; } x; x.f = f;
  unsigned short s = (unsigned short)((x.u + 0x7FFFu + ((x.u >> 16) & 1u)) >> 16);
  return __builtin_bit_cast(bf16_t, s);
}

// A fragment: 16x32 bf16 tile of row-major A, base points at (row0, k0).
// lane L: M = L%16 ; element e: K = e + (L/16)*16  -> 16 contiguous bf16/lane
__device__ __forceinline__ v16bf load_a(const bf16_t* base, int lda, int lane) {
  int m = lane & 15, g = (lane >> 4) & 1;
  const bf16_t* p = base + m * lda + g * 16;
  v16bf a;
#pragma unroll
  for (int e = 0; e < 16; ++e) a[e] = p[e];
  return a;
}

// B fragment: 32x16 (KxN) gathered from "in[n][k]" (contracted dim minor).
// lane L: N = L%16 ; element e: K = e + (L/16)*16  -> 16 contiguous bf16/lane
__device__ __forceinline__ v16bf load_b(const bf16_t* base, int ldb, int lane) {
  int n = lane & 15, g = (lane >> 4) & 1;
  const bf16_t* p = base + n * ldb + g * 16;
  v16bf b;
#pragma unroll
  for (int e = 0; e < 16; ++e) b[e] = p[e];
  return b;
}

// ---------------------------------------------------------------------------
// small utility kernels
// ---------------------------------------------------------------------------
__global__ void k_basis(bf16_t* bas) {
  int idx = blockIdx.x * blockDim.x + threadIdx.x;   // 65536
  int i = idx >> 8, j = idx & 255;
  float ang = (6.283185307179586f / 256.0f) * (float)((i * j) & 255);
  float c = __cosf(ang), s = __sinf(ang);
  bas[idx]              = f2bf(c);    // Cb = cos
  bas[idx + 65536]      = f2bf(s);    // Sb = +sin
  bas[idx + 131072]     = f2bf(-s);   // Nb = -sin
}

__global__ void k_f2bf(const float* __restrict__ s, bf16_t* __restrict__ d, int n) {
  for (int i = blockIdx.x * blockDim.x + threadIdx.x; i < n; i += gridDim.x * blockDim.x)
    d[i] = f2bf(s[i]);
}

__global__ void k_zero(float* p, int n) {
  for (int i = blockIdx.x * blockDim.x + threadIdx.x; i < n; i += gridDim.x * blockDim.x)
    p[i] = 0.0f;
}

// ---------------------------------------------------------------------------
// Pass 1: row DFT.  X[h][w] (bf16 plane) -> Frt[w'][h], Fit[w'][h]
//   F[h][w'] = sum_w X[h][w] * e^{-2pi i w w'/256}; matrices symmetric.
// per wave: plane = wave>>8 ; mt = w' tile ; nt = h tile
// ---------------------------------------------------------------------------
__global__ void k_rowfwd(const bf16_t* __restrict__ xb, const bf16_t* __restrict__ bas,
                         bf16_t* __restrict__ frt, bf16_t* __restrict__ fit) {
  int wave = (blockIdx.x * blockDim.x + threadIdx.x) >> 5;
  int lane = threadIdx.x & 31, g = lane >> 4, ln = lane & 15;
  int plane = wave >> 8, tile = wave & 255;
  int mt = tile >> 4, nt = tile & 15;
  const bf16_t* X  = xb + (size_t)plane * 65536;
  const bf16_t* Cb = bas;
  const bf16_t* Nb = bas + 131072;
  v8f ar = {}, ai = {};
  for (int k0 = 0; k0 < 256; k0 += 32) {
    v16bf b  = load_b(X + (nt * 16) * 256 + k0, 256, lane);
    v16bf ac = load_a(Cb + (mt * 16) * 256 + k0, 256, lane);
    v16bf an = load_a(Nb + (mt * 16) * 256 + k0, 256, lane);
    ar = wmma_bf16(ac, b, ar);
    ai = wmma_bf16(an, b, ai);
  }
  bf16_t* outr = frt + (size_t)plane * 65536;
  bf16_t* outi = fit + (size_t)plane * 65536;
#pragma unroll
  for (int r = 0; r < 8; ++r) {
    int m = mt * 16 + r + g * 8, n = nt * 16 + ln;   // m=w', n=h
    outr[m * 256 + n] = f2bf(ar[r]);
    outi[m * 256 + n] = f2bf(ai[r]);
  }
}

// ---------------------------------------------------------------------------
// Pass 2: column DFT + ortho scale -> fc[b][pix][c] (pix = h'*256+w', c minor)
//   Gr = C*Fr + S*Fi ; Gi = C*Fi - S*Fr   (S = +sin, applied via Sb/Nb)
// ---------------------------------------------------------------------------
__global__ void k_colfwd(const bf16_t* __restrict__ frt, const bf16_t* __restrict__ fit,
                         const bf16_t* __restrict__ bas, bf16_t* __restrict__ fc) {
  int wave = (blockIdx.x * blockDim.x + threadIdx.x) >> 5;
  int lane = threadIdx.x & 31, g = lane >> 4, ln = lane & 15;
  int plane = wave >> 8, tile = wave & 255;
  int mt = tile >> 4, nt = tile & 15;          // mt: h' tile, nt: w' tile
  const bf16_t* Fr = frt + (size_t)plane * 65536;   // [w'][h], contract h
  const bf16_t* Fi = fit + (size_t)plane * 65536;
  const bf16_t* Cb = bas;
  const bf16_t* Sb = bas + 65536;
  const bf16_t* Nb = bas + 131072;
  v8f gr = {}, gi = {};
  for (int k0 = 0; k0 < 256; k0 += 32) {
    v16bf br = load_b(Fr + (nt * 16) * 256 + k0, 256, lane);
    v16bf bi = load_b(Fi + (nt * 16) * 256 + k0, 256, lane);
    v16bf ac = load_a(Cb + (mt * 16) * 256 + k0, 256, lane);
    v16bf as = load_a(Sb + (mt * 16) * 256 + k0, 256, lane);
    v16bf an = load_a(Nb + (mt * 16) * 256 + k0, 256, lane);
    gr = wmma_bf16(ac, br, gr);
    gr = wmma_bf16(as, bi, gr);
    gi = wmma_bf16(ac, bi, gi);
    gi = wmma_bf16(an, br, gi);
  }
  int b = plane >> 6, c = plane & 63;
  bf16_t* fcb = fc + (size_t)b * 65536 * 128;
  const float sc = 1.0f / 256.0f;              // ortho 1/sqrt(65536)
#pragma unroll
  for (int r = 0; r < 8; ++r) {
    int hp = mt * 16 + r + g * 8, wp = nt * 16 + ln;
    int pix = hp * 256 + wp;
    fcb[(size_t)pix * 128 + c]      = f2bf(gr[r] * sc);
    fcb[(size_t)pix * 128 + 64 + c] = f2bf(gi[r] * sc);
  }
}

// ---------------------------------------------------------------------------
// Pass 3: Q,K conv + elu(+1) + per-head S += Q*K^T (sum over all pixels)
// grid (128, B), block 256 (8 waves). 32-pixel blocks staged through LDS.
// ---------------------------------------------------------------------------
__global__ void k_qk(const bf16_t* __restrict__ fc,
                     const bf16_t* __restrict__ wqb, const float* __restrict__ bq,
                     const bf16_t* __restrict__ wkb, const float* __restrict__ bk,
                     float* __restrict__ Sg) {
  __shared__ bf16_t Ql[128 * 32];
  __shared__ bf16_t Kl[128 * 32];
  int b = blockIdx.y;
  int wave = threadIdx.x >> 5, lane = threadIdx.x & 31;
  int g = lane >> 4, ln = lane & 15;
  const bf16_t* fcb = fc + (size_t)b * 65536 * 128;
  v8f s0 = {}, s1 = {};
  const int t0 = wave * 2, t1 = wave * 2 + 1;   // 16 S-tiles over 8 waves

  for (int blk = blockIdx.x; blk < 2048; blk += gridDim.x) {
    int pix0 = blk * 32;
    // ---- compute Q,K for channels [16*wave, 16*wave+16), pixels [pix0,pix0+32)
    v8f q[2], kk[2];
#pragma unroll
    for (int pt = 0; pt < 2; ++pt) {
      v8f aq = {}, ak = {};
      for (int k0 = 0; k0 < 128; k0 += 32) {
        v16bf bfrag = load_b(fcb + (size_t)(pix0 + pt * 16) * 128 + k0, 128, lane);
        v16bf aqf = load_a(wqb + (wave * 16) * 128 + k0, 128, lane);
        v16bf akf = load_a(wkb + (wave * 16) * 128 + k0, 128, lane);
        aq = wmma_bf16(aqf, bfrag, aq);
        ak = wmma_bf16(akf, bfrag, ak);
      }
      q[pt] = aq; kk[pt] = ak;
    }
    // ---- bias + (elu+1) -> LDS [chan][pix], pitch 32
#pragma unroll
    for (int pt = 0; pt < 2; ++pt) {
#pragma unroll
      for (int r = 0; r < 8; ++r) {
        int chan = wave * 16 + r + g * 8;
        int pix  = pt * 16 + ln;
        float qv = q[pt][r] + bq[chan];
        qv = (qv > 0.0f) ? (qv + 1.0f) : __expf(qv);   // elu(x)+1
        float kv = kk[pt][r] + bk[chan];
        kv = (kv > 0.0f) ? (kv + 1.0f) : __expf(kv);
        Ql[chan * 32 + pix] = f2bf(qv);
        Kl[chan * 32 + pix] = f2bf(kv);
      }
    }
    __syncthreads();
    // ---- S tiles: tile t -> head=t>>2, mi=(t>>1)&1, ni=t&1 ; K-dim = 32 pixels
    {
      int t = t0, head = t >> 2, mi = (t >> 1) & 1, ni = t & 1;
      v16bf af  = load_a(Ql + (head * 32 + mi * 16) * 32, 32, lane);
      v16bf bfg = load_b(Kl + (head * 32 + ni * 16) * 32, 32, lane);
      s0 = wmma_bf16(af, bfg, s0);
      t = t1; head = t >> 2; mi = (t >> 1) & 1; ni = t & 1;
      af  = load_a(Ql + (head * 32 + mi * 16) * 32, 32, lane);
      bfg = load_b(Kl + (head * 32 + ni * 16) * 32, 32, lane);
      s1 = wmma_bf16(af, bfg, s1);
    }
    __syncthreads();
  }
  float* Sb_ = Sg + b * 4096;
  {
    int t = t0, head = t >> 2, mi = (t >> 1) & 1, ni = t & 1;
#pragma unroll
    for (int r = 0; r < 8; ++r) {
      int m = mi * 16 + r + g * 8, n = ni * 16 + ln;
      atomicAdd(&Sb_[head * 1024 + m * 32 + n], s0[r]);
    }
    t = t1; head = t >> 2; mi = (t >> 1) & 1; ni = t & 1;
#pragma unroll
    for (int r = 0; r < 8; ++r) {
      int m = mi * 16 + r + g * 8, n = ni * 16 + ln;
      atomicAdd(&Sb_[head * 1024 + m * 32 + n], s1[r]);
    }
  }
}

// ---------------------------------------------------------------------------
// Pass 4: M_b = wo * S_blockdiag * wv (bf16), cvec_b = wo*(S_bd*bv) + bo (f32)
// one block per batch; T staged through global scratch TB.
// ---------------------------------------------------------------------------
__global__ void k_combine(const float* __restrict__ Sg,
                          const float* __restrict__ wv, const float* __restrict__ bv,
                          const float* __restrict__ wo, const float* __restrict__ bo,
                          bf16_t* __restrict__ Mb, float* __restrict__ cvec,
                          float* __restrict__ TB) {
  __shared__ float Sl[4096];
  __shared__ float bvt[128];
  int b = blockIdx.x, tid = threadIdx.x;
  float* Tb = TB + b * 16384;
  for (int i = tid; i < 4096; i += 256) Sl[i] = Sg[b * 4096 + i];
  __syncthreads();
  for (int i = tid; i < 16384; i += 256) {       // T[d][j] = sum_e S[h][dd][e] wv[h*32+e][j]
    int d = i >> 7, j = i & 127, head = d >> 5, dd = d & 31;
    float acc = 0.0f;
    for (int e = 0; e < 32; ++e)
      acc += Sl[head * 1024 + dd * 32 + e] * wv[(head * 32 + e) * 128 + j];
    Tb[i] = acc;
  }
  if (tid < 128) {
    int d = tid, head = d >> 5, dd = d & 31;
    float acc = 0.0f;
    for (int e = 0; e < 32; ++e) acc += Sl[head * 1024 + dd * 32 + e] * bv[head * 32 + e];
    bvt[d] = acc;
  }
  __syncthreads();
  for (int i = tid; i < 16384; i += 256) {       // Mb[o][j] = sum_d wo[o][d] T[d][j]
    int o = i >> 7, j = i & 127;
    float acc = 0.0f;
    for (int d = 0; d < 128; ++d) acc += wo[o * 128 + d] * Tb[d * 128 + j];
    Mb[b * 16384 + i] = f2bf(acc);
  }
  if (tid < 128) {
    int o = tid;
    float acc = bo[o];
    for (int d = 0; d < 128; ++d) acc += wo[o * 128 + d] * bvt[d];
    cvec[b * 128 + o] = acc;
  }
}

// ---------------------------------------------------------------------------
// Pass 5: op[b][:,pix] = M_b * fc[b][:,pix] + cvec_b -> opr/opi planes [h'][w']
// ---------------------------------------------------------------------------
__global__ void k_op(const bf16_t* __restrict__ fc, const bf16_t* __restrict__ Mb,
                     const float* __restrict__ cvec,
                     bf16_t* __restrict__ opr, bf16_t* __restrict__ opi) {
  int wave = (blockIdx.x * blockDim.x + threadIdx.x) >> 5;
  int lane = threadIdx.x & 31, g = lane >> 4, ln = lane & 15;
  int nt = wave & 4095, mt = (wave >> 12) & 7, b = wave >> 15;
  const bf16_t* fcb = fc + (size_t)b * 65536 * 128;
  const bf16_t* A = Mb + b * 16384;
  v8f acc = {};
  for (int k0 = 0; k0 < 128; k0 += 32) {
    v16bf bf_ = load_b(fcb + (size_t)(nt * 16) * 128 + k0, 128, lane);
    v16bf af  = load_a(A + (mt * 16) * 128 + k0, 128, lane);
    acc = wmma_bf16(af, bf_, acc);
  }
#pragma unroll
  for (int r = 0; r < 8; ++r) {
    int chan = mt * 16 + r + g * 8;
    int pix  = nt * 16 + ln;
    float v = acc[r] + cvec[b * 128 + chan];
    if (chan < 64) opr[(((size_t)(b * 64 + chan)) << 16) + pix] = f2bf(v);
    else           opi[(((size_t)(b * 64 + chan - 64)) << 16) + pix] = f2bf(v);
  }
}

// ---------------------------------------------------------------------------
// Pass 6: inverse row transform (contract w'): U[h'][w] stored as Ut[w][h']
//   Ur = C*OPr - S*OPi ; Ui = S*OPr + C*OPi   (e^{+i theta})
// ---------------------------------------------------------------------------
__global__ void k_invrow(const bf16_t* __restrict__ opr, const bf16_t* __restrict__ opi,
                         const bf16_t* __restrict__ bas,
                         bf16_t* __restrict__ urt, bf16_t* __restrict__ uit) {
  int wave = (blockIdx.x * blockDim.x + threadIdx.x) >> 5;
  int lane = threadIdx.x & 31, g = lane >> 4, ln = lane & 15;
  int plane = wave >> 8, tile = wave & 255;
  int mt = tile >> 4, nt = tile & 15;          // mt: w tile, nt: h' tile
  const bf16_t* Pr = opr + (size_t)plane * 65536;   // [h'][w'], contract w'
  const bf16_t* Pi = opi + (size_t)plane * 65536;
  const bf16_t* Cb = bas;
  const bf16_t* Sb = bas + 65536;
  const bf16_t* Nb = bas + 131072;
  v8f ur = {}, ui = {};
  for (int k0 = 0; k0 < 256; k0 += 32) {
    v16bf br = load_b(Pr + (nt * 16) * 256 + k0, 256, lane);
    v16bf bi = load_b(Pi + (nt * 16) * 256 + k0, 256, lane);
    v16bf ac = load_a(Cb + (mt * 16) * 256 + k0, 256, lane);
    v16bf as = load_a(Sb + (mt * 16) * 256 + k0, 256, lane);
    v16bf an = load_a(Nb + (mt * 16) * 256 + k0, 256, lane);
    ur = wmma_bf16(ac, br, ur);
    ur = wmma_bf16(an, bi, ur);
    ui = wmma_bf16(ac, bi, ui);
    ui = wmma_bf16(as, br, ui);
  }
  bf16_t* Or = urt + (size_t)plane * 65536;
  bf16_t* Oi = uit + (size_t)plane * 65536;
#pragma unroll
  for (int r = 0; r < 8; ++r) {
    int m = mt * 16 + r + g * 8, n = nt * 16 + ln;   // m=w, n=h'
    Or[m * 256 + n] = f2bf(ur[r]);
    Oi[m * 256 + n] = f2bf(ui[r]);
  }
}

// ---------------------------------------------------------------------------
// Pass 7: inverse col transform (contract h'), keep real part, ortho scale.
//   y[h][w] = (C*Ur - S*Ui)/256 -> y[b][pix][c] (c minor) for final conv
// ---------------------------------------------------------------------------
__global__ void k_invcol(const bf16_t* __restrict__ urt, const bf16_t* __restrict__ uit,
                         const bf16_t* __restrict__ bas, bf16_t* __restrict__ y) {
  int wave = (blockIdx.x * blockDim.x + threadIdx.x) >> 5;
  int lane = threadIdx.x & 31, g = lane >> 4, ln = lane & 15;
  int plane = wave >> 8, tile = wave & 255;
  int mt = tile >> 4, nt = tile & 15;          // mt: h tile, nt: w tile
  const bf16_t* Ur = urt + (size_t)plane * 65536;   // [w][h'], contract h'
  const bf16_t* Ui = uit + (size_t)plane * 65536;
  const bf16_t* Cb = bas;
  const bf16_t* Nb = bas + 131072;
  v8f acc = {};
  for (int k0 = 0; k0 < 256; k0 += 32) {
    v16bf br = load_b(Ur + (nt * 16) * 256 + k0, 256, lane);
    v16bf bi = load_b(Ui + (nt * 16) * 256 + k0, 256, lane);
    v16bf ac = load_a(Cb + (mt * 16) * 256 + k0, 256, lane);
    v16bf an = load_a(Nb + (mt * 16) * 256 + k0, 256, lane);
    acc = wmma_bf16(ac, br, acc);
    acc = wmma_bf16(an, bi, acc);
  }
  int b = plane >> 6, c = plane & 63;
  bf16_t* yb = y + (size_t)b * 65536 * 64;
  const float sc = 1.0f / 256.0f;
#pragma unroll
  for (int r = 0; r < 8; ++r) {
    int h = mt * 16 + r + g * 8, w = nt * 16 + ln;
    yb[(size_t)(h * 256 + w) * 64 + c] = f2bf(acc[r] * sc);
  }
}

// ---------------------------------------------------------------------------
// Pass 8: s = sigmoid(ws*y + bs) -> d_out (f32, standard BCHW layout)
// ---------------------------------------------------------------------------
__global__ void k_final(const bf16_t* __restrict__ y, const bf16_t* __restrict__ wsb,
                        const float* __restrict__ bs, float* __restrict__ out) {
  int wave = (blockIdx.x * blockDim.x + threadIdx.x) >> 5;
  int lane = threadIdx.x & 31, g = lane >> 4, ln = lane & 15;
  int nt = wave & 4095, mt = (wave >> 12) & 3, b = wave >> 14;
  const bf16_t* yb = y + (size_t)b * 65536 * 64;
  v8f acc = {};
  for (int k0 = 0; k0 < 64; k0 += 32) {
    v16bf bf_ = load_b(yb + (size_t)(nt * 16) * 64 + k0, 64, lane);
    v16bf af  = load_a(wsb + (mt * 16) * 64 + k0, 64, lane);
    acc = wmma_bf16(af, bf_, acc);
  }
#pragma unroll
  for (int r = 0; r < 8; ++r) {
    int co = mt * 16 + r + g * 8;
    int pix = nt * 16 + ln;
    float v = acc[r] + bs[co];
    out[(((size_t)(b * 64 + co)) << 16) + pix] = 1.0f / (1.0f + __expf(-v));
  }
}

// ---------------------------------------------------------------------------
extern "C" void kernel_launch(void* const* d_in, const int* in_sizes, int n_in,
                              void* d_out, int out_size, void* d_ws, size_t ws_size,
                              hipStream_t stream) {
  (void)in_sizes; (void)n_in; (void)out_size; (void)ws_size;
  const float* x   = (const float*)d_in[0];
  const float* wq  = (const float*)d_in[1];
  const float* bq  = (const float*)d_in[2];
  const float* wk  = (const float*)d_in[3];
  const float* bk  = (const float*)d_in[4];
  const float* wv  = (const float*)d_in[5];
  const float* bv  = (const float*)d_in[6];
  const float* wo  = (const float*)d_in[7];
  const float* bo  = (const float*)d_in[8];
  const float* wsm = (const float*)d_in[9];
  const float* bs  = (const float*)d_in[10];

  char* W = (char*)d_ws;
  // workspace layout (bytes); total ~161 MiB
  bf16_t* BAS = (bf16_t*)(W);                                  // 393216
  bf16_t* XB  = (bf16_t*)(W + 393216);                         // 32 MiB (reused as Y)
  bf16_t* FR  = (bf16_t*)(W + 393216 + 33554432);              // 32 MiB (reused as OPR)
  bf16_t* FI  = (bf16_t*)(W + 393216 + 2 * 33554432);          // 32 MiB (reused as OPI)
  bf16_t* FC  = (bf16_t*)(W + 393216 + 3 * 33554432);          // 64 MiB (reused URT/UIT)
  float*  SG  = (float*) (W + 393216 + 3 * 33554432 + 67108864);           // 64 KiB
  bf16_t* MB  = (bf16_t*)(W + 393216 + 3 * 33554432 + 67108864 + 65536);   // 128 KiB
  float*  CV  = (float*) (W + 393216 + 3 * 33554432 + 67108864 + 196608);  // 2 KiB
  float*  TB  = (float*) (W + 393216 + 3 * 33554432 + 67108864 + 198656);  // 256 KiB
  bf16_t* WQB = (bf16_t*)(W + 393216 + 3 * 33554432 + 67108864 + 460800);  // 32 KiB
  bf16_t* WKB = (bf16_t*)(W + 393216 + 3 * 33554432 + 67108864 + 493568);  // 32 KiB
  bf16_t* WSB = (bf16_t*)(W + 393216 + 3 * 33554432 + 67108864 + 526336);  // 8 KiB
  bf16_t* URT = FC;                                            // fc dead after k_op
  bf16_t* UIT = (bf16_t*)((char*)FC + 33554432);
  bf16_t* YB  = XB;                                            // xb dead after rowfwd

  k_basis <<<256, 256, 0, stream>>>(BAS);
  k_f2bf  <<<4096, 256, 0, stream>>>(x, XB, 16777216);
  k_f2bf  <<<64, 256, 0, stream>>>(wq, WQB, 16384);
  k_f2bf  <<<64, 256, 0, stream>>>(wk, WKB, 16384);
  k_f2bf  <<<16, 256, 0, stream>>>(wsm, WSB, 4096);
  k_zero  <<<64, 256, 0, stream>>>(SG, 16384);

  k_rowfwd<<<8192, 256, 0, stream>>>(XB, BAS, FR, FI);
  k_colfwd<<<8192, 256, 0, stream>>>(FR, FI, BAS, FC);
  k_qk    <<<dim3(128, 4), 256, 0, stream>>>(FC, WQB, bq, WKB, bk, SG);
  k_combine<<<4, 256, 0, stream>>>(SG, wv, bv, wo, bo, MB, CV, TB);
  k_op    <<<16384, 256, 0, stream>>>(FC, MB, CV, FR, FI);     // OPR=FR, OPI=FI
  k_invrow<<<8192, 256, 0, stream>>>(FR, FI, BAS, URT, UIT);
  k_invcol<<<8192, 256, 0, stream>>>(URT, UIT, BAS, YB);
  k_final <<<8192, 256, 0, stream>>>(YB, WSB, bs, (float*)d_out);
}
